// SparseConvMIL_60979945669307
// MI455X (gfx1250) — matrix operand order repaired and checked by reading, not compiled, not run on hardware
//
#include <hip/hip_runtime.h>
#include <hip/hip_bf16.h>

// ---------------------------------------------------------------------------
// Problem constants (match reference)
// ---------------------------------------------------------------------------
#define BB    16          // bags
#define NPTS  4096        // points per bag
#define CIN   128
#define CHID  128
#define NCLS  10
#define GG    128         // grid extent (4096/32)
#define DS_SHIFT 5        // downsample = 32 -> >>5
#define CELLS (BB*GG*GG)  // 262144

typedef __attribute__((ext_vector_type(16))) __bf16 v16bf;
typedef __attribute__((ext_vector_type(8)))  float  v8f;

union BF16x16 { uint4 q[2]; v16bf v; };

#if defined(__gfx1250__) && __has_builtin(__builtin_amdgcn_global_load_async_to_lds_b128) && __has_builtin(__builtin_amdgcn_s_wait_asynccnt)
#define HAVE_ASYNC_LDS 1
typedef int v4i_a __attribute__((vector_size(4 * sizeof(int))));
typedef __attribute__((address_space(1))) v4i_a GV4;   // global int4*
typedef __attribute__((address_space(3))) v4i_a LV4;   // LDS int4*
#else
#define HAVE_ASYNC_LDS 0
#endif

__device__ __forceinline__ unsigned short f2bf(float f) {
    unsigned int u = __float_as_uint(f);
    unsigned int r = u + 0x7FFFu + ((u >> 16) & 1u);   // RNE
    return (unsigned short)(r >> 16);
}
__device__ __forceinline__ float bf2f(unsigned short h) {
    return __uint_as_float(((unsigned int)h) << 16);
}
__device__ __forceinline__ unsigned int pack2bf(float a, float b) {
    return (unsigned int)f2bf(a) | ((unsigned int)f2bf(b) << 16);
}

// ---------------------------------------------------------------------------
// 1) zero the atomic accumulators (counts || sums, contiguous)
// ---------------------------------------------------------------------------
__global__ void k_zero_f32(float* __restrict__ p, long n) {
    long i = (long)blockIdx.x * blockDim.x + threadIdx.x;
    long stride = (long)gridDim.x * blockDim.x;
    for (; i < n; i += stride) p[i] = 0.0f;
}

// ---------------------------------------------------------------------------
// 2) scatter points into per-cell sums + counts (float atomics)
// ---------------------------------------------------------------------------
__global__ void k_scatter(const float* __restrict__ x,
                          const int*   __restrict__ coords,
                          float* __restrict__ sums,
                          float* __restrict__ counts) {
    int p = blockIdx.x * 2 + (threadIdx.x >> 7);
    int c = threadIdx.x & 127;
    int b = p >> 12;                                   // p / 4096
    int cx = coords[(size_t)p * 2 + 0];
    int cy = coords[(size_t)p * 2 + 1];
    int qx = cx >> DS_SHIFT;
    int qy = cy >> DS_SHIFT;
    int cell = b * GG * GG + qx * GG + qy;
    atomicAdd(&sums[(size_t)cell * CIN + c], x[(size_t)p * CIN + c]);
    if (c == 0) atomicAdd(&counts[cell], 1.0f);
}

// ---------------------------------------------------------------------------
// 3) normalize: grid_bf16 = sums / max(count,1)   (8 bf16 per store)
// ---------------------------------------------------------------------------
__global__ void k_normalize(const float4* __restrict__ sums4,
                            const float*  __restrict__ counts,
                            uint4* __restrict__ grid4) {
    long n = (long)CELLS * (CIN / 8);                  // groups of 8 channels
    long i = (long)blockIdx.x * blockDim.x + threadIdx.x;
    long stride = (long)gridDim.x * blockDim.x;
    for (; i < n; i += stride) {
        float inv = 1.0f / fmaxf(counts[i >> 4], 1.0f);
        float4 a = sums4[i * 2 + 0];
        float4 b = sums4[i * 2 + 1];
        uint4 o;
        o.x = pack2bf(a.x * inv, a.y * inv);
        o.y = pack2bf(a.z * inv, a.w * inv);
        o.z = pack2bf(b.x * inv, b.y * inv);
        o.w = pack2bf(b.z * inv, b.w * inv);
        grid4[i] = o;
    }
}

// ---------------------------------------------------------------------------
// 4) pack conv weights f32 [3,3,128,128] -> bf16 B-fragment layout
//    packed[((ntile*9 + tap)*4 + kc)*512 + lane*16 + e]
//    with K = e + 16*(lane>=16), N = lane&15 (contiguous K order, same as the A loads)
// ---------------------------------------------------------------------------
__global__ void k_pack_weights(const float* __restrict__ W,
                               unsigned short* __restrict__ packed) {
    int idx = blockIdx.x * blockDim.x + threadIdx.x;   // 8*9*4*32*16 = 147456
    if (idx >= 8 * 9 * 4 * 32 * 16) return;
    int e     = idx & 15;
    int t1    = idx >> 4;
    int lane  = t1 & 31;
    int t2    = t1 >> 5;
    int kc    = t2 & 3;
    int t3    = t2 >> 2;
    int tap   = t3 % 9;
    int ntile = t3 / 9;
    int K     = e + ((lane & 16) ? 16 : 0);
    int cin   = kc * 32 + K;
    int cout  = ntile * 16 + (lane & 15);
    float w = W[((size_t)tap * CIN + cin) * CHID + cout];
    packed[idx] = f2bf(w);
}

// ---------------------------------------------------------------------------
// 5) implicit-GEMM 3x3 conv via v_wmma_f32_16x16x32_bf16
//    block = (b, y, 32-wide x tile); 8 waves = 8 cout tiles of 16,
//    each wave computes TWO 16x16 tiles sharing one B fragment.
// ---------------------------------------------------------------------------
#define HALO_W 34
__global__ void __launch_bounds__(256)
k_conv3x3_wmma(const unsigned short* __restrict__ gin,
               const unsigned short* __restrict__ wpk,
               const float* __restrict__ counts,
               unsigned short* __restrict__ gout) {
    __shared__ unsigned short hs[3 * HALO_W * CIN];    // halo: 26112 B
    __shared__ float msk[32];

    const int tid = threadIdx.x;
    const int blk = blockIdx.x;                        // 8192 blocks
    const int b   = blk >> 9;                          // 512 blocks / batch
    const int rem = blk & 511;
    const int y   = rem >> 2;
    const int x0  = (rem & 3) << 5;

    // cooperative halo fill (16 B chunks); zero edges, async interior
    for (int ch = tid; ch < 3 * HALO_W * 16; ch += 256) {
        int c8 = ch & 15;
        int cc = (ch >> 4) % HALO_W;
        int rr = (ch >> 4) / HALO_W;
        int gy = y + rr - 1;
        int gx = x0 + cc - 1;
        unsigned short* lp = hs + (((rr * HALO_W + cc)) << 7) + (c8 << 3);
        if (gy >= 0 && gy < GG && gx >= 0 && gx < GG) {
            const unsigned short* gp =
                gin + ((((size_t)b * GG + gy) * GG + gx) << 7) + (c8 << 3);
#if HAVE_ASYNC_LDS
            __builtin_amdgcn_global_load_async_to_lds_b128((GV4*)gp, (LV4*)lp, 0, 0);
#else
            *(uint4*)lp = *(const uint4*)gp;
#endif
        } else {
            *(uint4*)lp = make_uint4(0u, 0u, 0u, 0u);
        }
    }
    if (tid < 32) {
        float c = counts[((size_t)b * GG + y) * GG + x0 + tid];
        msk[tid] = (c > 0.0f) ? 1.0f : 0.0f;
    }
#if HAVE_ASYNC_LDS
    __builtin_amdgcn_s_wait_asynccnt(0);
#endif
    __syncthreads();

    const int lane = tid & 31;
    const int w    = tid >> 5;                         // cout tile 0..7
    const int m    = lane & 15;                        // spatial row of A
    const int hiA  = (lane & 16) ? 16 : 0;             // A-layout K base

    v8f acc0 = {0.f, 0.f, 0.f, 0.f, 0.f, 0.f, 0.f, 0.f};
    v8f acc1 = {0.f, 0.f, 0.f, 0.f, 0.f, 0.f, 0.f, 0.f};

    const unsigned short* wbase = wpk + (size_t)w * 9 * 4 * 512 + (size_t)lane * 16;
    __builtin_prefetch(wbase, 0, 1);

    for (int tap = 0; tap < 9; ++tap) {
        const int dy = tap / 3, dx = tap % 3;
        const unsigned short* arow0 = hs + (((dy * HALO_W + m + dx)) << 7) + hiA;
        const unsigned short* arow1 = arow0 + (16 << 7);      // x + 16
        const unsigned short* brow  = wbase + (size_t)tap * 4 * 512;
#pragma unroll
        for (int kc = 0; kc < 4; ++kc) {
            BF16x16 a0, a1, bm;
            bm.q[0] = *(const uint4*)(brow + kc * 512);
            bm.q[1] = *(const uint4*)(brow + kc * 512 + 8);
            a0.q[0] = *(const uint4*)(arow0 + kc * 32);       // K = hiA .. hiA+7
            a0.q[1] = *(const uint4*)(arow0 + kc * 32 + 8);   // K = hiA+8 .. +15
            a1.q[0] = *(const uint4*)(arow1 + kc * 32);
            a1.q[1] = *(const uint4*)(arow1 + kc * 32 + 8);
            acc0 = __builtin_amdgcn_wmma_f32_16x16x32_bf16(
                false, a0.v, false, bm.v, (short)0, acc0, false, false);
            acc1 = __builtin_amdgcn_wmma_f32_16x16x32_bf16(
                false, a1.v, false, bm.v, (short)0, acc1, false, false);
        }
    }

    // C/D layout: M = r + 8*(lane>=16), N = lane&15
    const int mrow = (lane & 16) ? 8 : 0;
    const int n    = w * 16 + (lane & 15);
    const size_t rowbase = (((size_t)b * GG + y) * GG + x0) * CHID + n;
#pragma unroll
    for (int r = 0; r < 8; ++r) {
        int M0 = mrow + r;
        int M1 = 16 + mrow + r;
        float v0 = fmaxf(acc0[r], 0.0f) * msk[M0];
        float v1 = fmaxf(acc1[r], 0.0f) * msk[M1];
        gout[rowbase + (size_t)M0 * CHID] = f2bf(v0);
        gout[rowbase + (size_t)M1 * CHID] = f2bf(v1);
    }
}

// ---------------------------------------------------------------------------
// 6) masked global-average pool + classifier (one block per bag)
// ---------------------------------------------------------------------------
__global__ void __launch_bounds__(256)
k_pool_classify(const unsigned short* __restrict__ h2,
                const float* __restrict__ counts,
                const float* __restrict__ Wc,
                const float* __restrict__ bc,
                float* __restrict__ out) {
    __shared__ float ps[CHID];
    __shared__ float np2[2];

    const int b    = blockIdx.x;
    const int tid  = threadIdx.x;
    const int c    = tid & 127;
    const int half = tid >> 7;

    float acc = 0.0f;
    float occ = 0.0f;
    for (int i = half; i < GG * GG; i += 2) {
        float cnt = counts[(size_t)b * GG * GG + i];
        if (cnt > 0.0f) {
            acc += bf2f(h2[((size_t)b * GG * GG + i) * CHID + c]);
            occ += 1.0f;
        }
    }
    if (half == 0) ps[c] = acc;
    if (c == 0) np2[half] = occ;
    __syncthreads();
    if (half == 1) ps[c] += acc;
    __syncthreads();

    if (tid < NCLS) {
        float npts = np2[0] + np2[1];
        float l = bc[tid];
        for (int cc = 0; cc < CHID; ++cc)
            l += (ps[cc] / npts) * Wc[(size_t)cc * NCLS + tid];
        out[(size_t)b * NCLS + tid] = l;
    }
}

// ---------------------------------------------------------------------------
// launch
// ---------------------------------------------------------------------------
extern "C" void kernel_launch(void* const* d_in, const int* in_sizes, int n_in,
                              void* d_out, int out_size, void* d_ws, size_t ws_size,
                              hipStream_t stream) {
    const float* x      = (const float*)d_in[0];
    const int*   coords = (const int*)d_in[1];
    const float* W1     = (const float*)d_in[2];
    const float* W2     = (const float*)d_in[3];
    const float* Wc     = (const float*)d_in[4];
    const float* bc     = (const float*)d_in[5];
    float* logits       = (float*)d_out;

    char* ws = (char*)d_ws;
    const size_t S_COUNTS = (size_t)CELLS * 4;              //   1 MB
    const size_t S_SUMS   = (size_t)CELLS * CIN * 4;        // 128 MB
    const size_t S_GRID   = (size_t)CELLS * CIN * 2;        //  64 MB
    const size_t S_WPK    = (size_t)8 * 9 * 4 * 32 * 16 * 2;// 288 KB

    float*          counts = (float*)(ws);
    float*          sums   = (float*)(ws + S_COUNTS);
    unsigned short* grid   = (unsigned short*)(ws + S_COUNTS + S_SUMS);
    unsigned short* h1     = (unsigned short*)(ws + S_COUNTS + S_SUMS + S_GRID);
    unsigned short* w1p    = (unsigned short*)(ws + S_COUNTS + S_SUMS + 2 * S_GRID);
    unsigned short* w2p    = (unsigned short*)(ws + S_COUNTS + S_SUMS + 2 * S_GRID + S_WPK);
    unsigned short* h2     = (unsigned short*)(ws + S_COUNTS);   // alias dead sums buffer

    // 1) zero counts+sums (contiguous)
    long nzero = (long)CELLS + (long)CELLS * CIN;
    k_zero_f32<<<2048, 256, 0, stream>>>(counts, nzero);

    // 4) pack weights
    k_pack_weights<<<(8 * 9 * 4 * 32 * 16 + 255) / 256, 256, 0, stream>>>(W1, w1p);
    k_pack_weights<<<(8 * 9 * 4 * 32 * 16 + 255) / 256, 256, 0, stream>>>(W2, w2p);

    // 2) scatter
    k_scatter<<<(BB * NPTS) / 2, 256, 0, stream>>>(x, coords, sums, counts);

    // 3) normalize -> bf16 grid
    k_normalize<<<2048, 256, 0, stream>>>((const float4*)sums, counts, (uint4*)grid);

    // 5) conv1: grid -> h1 ; conv2: h1 -> h2
    const int conv_blocks = BB * GG * (GG / 32);             // 8192
    k_conv3x3_wmma<<<conv_blocks, 256, 0, stream>>>(grid, w1p, counts, h1);
    k_conv3x3_wmma<<<conv_blocks, 256, 0, stream>>>(h1, w2p, counts, h2);

    // 6) pool + classifier
    k_pool_classify<<<BB, 256, 0, stream>>>(h2, counts, Wc, bc, logits);
}
